// SelfAttention_79499844649017
// MI455X (gfx1250) — matrix operand order repaired and checked by reading, not compiled, not run on hardware
//
#include <hip/hip_runtime.h>

typedef __attribute__((ext_vector_type(16))) _Float16 v16h;
typedef __attribute__((ext_vector_type(8)))  float    v8f;

#define P_ 8
#define N_ 1024
#define D_ 512
#define M_ 8
#define C_ 64
#define PN_ 8192
#define LOG2E 1.44269504088896f

__device__ __forceinline__ v8f zero8() {
    v8f r = {0.f, 0.f, 0.f, 0.f, 0.f, 0.f, 0.f, 0.f};
    return r;
}

// WMMA wrapper: D = A(16x32 f16) * B(32x16 f16) + C(16x16 f32)
__device__ __forceinline__ v8f wmma_f16(v16h a, v16h b, v8f c) {
    return __builtin_amdgcn_wmma_f32_16x16x32_f16(
        false, a, false, b, (short)0, c, false, false);
}

// ---- CDNA5 async global->LDS copy (ASYNCcnt path) ----
// 16B per lane: LDS dest offset = low 32 bits of generic pointer (ISA flat->LDS truncation).
__device__ __forceinline__ void async_copy_b128(const void* gsrc, void* ldst) {
    unsigned lds = (unsigned)(unsigned long long)(uintptr_t)ldst;
    unsigned long long g = (unsigned long long)(uintptr_t)gsrc;
    asm volatile("global_load_async_to_lds_b128 %0, %1, off"
                 :: "v"(lds), "v"(g) : "memory");
}
__device__ __forceinline__ void wait_async_le8() {
    asm volatile("s_wait_asynccnt 0x8" ::: "memory");
}
__device__ __forceinline__ void wait_async_0() {
    asm volatile("s_wait_asynccnt 0x0" ::: "memory");
}

// A fragment (16 rows x 32 K), source row-major with row stride `stride`.
// ISA layout: lane = row (lane&15); lanes<16 hold K{0..7,16..23}, lanes>=16 K{8..15,24..31}.
__device__ __forceinline__ v16h load_frag_a(const _Float16* base, int stride, int lane) {
    const int r    = lane & 15;
    const int koff = (lane & 16) ? 8 : 0;
    const _Float16* p0 = base + r * stride + koff;
    union { v16h h; uint4 u[2]; } f;
    f.u[0] = *reinterpret_cast<const uint4*>(p0);
    f.u[1] = *reinterpret_cast<const uint4*>(p0 + 16);
    return f.h;
}

// B fragment (32 K x 16 cols): element(k,n) at base + n*stride + k.
// ISA layout: lane = col (lane&15); lanes<16 hold K{0..7,16..23}, lanes>=16 K{8..15,24..31}.
__device__ __forceinline__ v16h load_frag_b(const _Float16* base, int stride, int lane) {
    const int n    = lane & 15;
    const int koff = (lane & 16) ? 8 : 0;
    const _Float16* p0 = base + n * stride + koff;
    union { v16h h; uint4 u[2]; } f;
    f.u[0] = *reinterpret_cast<const uint4*>(p0);
    f.u[1] = *reinterpret_cast<const uint4*>(p0 + 16);
    return f.h;
}

// ---------------- f32 -> f16 convert ----------------
__global__ void cvt_f32_f16_kernel(const float* __restrict__ s, _Float16* __restrict__ d, int n) {
    int i = blockIdx.x * 256 + threadIdx.x;
    if (i < n) d[i] = (_Float16)s[i];
}

// ---------------- fused QKV projection ----------------
// h16 [PN, D] @ W{q,k,v}^T. 8 waves per block share one 16-row A tile, staged in LDS
// via async copies. Head-split scatter: Q,K -> [m,p,n,c] (Q pre-scaled 0.125), V -> [m,p,c,n].
__global__ __launch_bounds__(256) void qkv_kernel(
    const _Float16* __restrict__ h16,
    const _Float16* __restrict__ Wq, const _Float16* __restrict__ Wk, const _Float16* __restrict__ Wv,
    const float* __restrict__ bq, const float* __restrict__ bk, const float* __restrict__ bv,
    _Float16* __restrict__ Qh, _Float16* __restrict__ Kh, _Float16* __restrict__ Vt)
{
    __shared__ __align__(16) _Float16 ldsA[16 * D_];   // 16 KB: shared A row tile
    const int wid  = blockIdx.x * 8 + (threadIdx.x >> 5);
    const int lane = threadIdx.x & 31;
    const int rt = wid >> 5;   // 512 row tiles of 16 (uniform within block: 8 | 32)
    const int ct = wid & 31;   // 32 col tiles of 16
    // stage A tile (16 x 512 f16 = 1024 uint4) with async copies
    {
        const uint4* src = reinterpret_cast<const uint4*>(h16 + (size_t)rt * 16 * D_);
        uint4* dst = reinterpret_cast<uint4*>(ldsA);
        for (int i = 0; i < 4; ++i) {
            const int idx = threadIdx.x + i * 256;
            async_copy_b128(src + idx, dst + idx);
        }
    }
    wait_async_0();
    __syncthreads();

    v8f aq = zero8(), ak = zero8(), av = zero8();
    const size_t wrow = (size_t)ct * 16 * D_;
    for (int k0 = 0; k0 < D_; k0 += 32) {
        v16h a = load_frag_a(ldsA + k0, D_, lane);
        aq = wmma_f16(a, load_frag_b(Wq + wrow + k0, D_, lane), aq);
        ak = wmma_f16(a, load_frag_b(Wk + wrow + k0, D_, lane), ak);
        av = wmma_f16(a, load_frag_b(Wv + wrow + k0, D_, lane), av);
    }
    const int d = ct * 16 + (lane & 15);
    const int m = d & 7;        // head index INNERMOST in feature dim (d = c*M + m)
    const int c = d >> 3;
    const float bqv = bq[d], bkv = bk[d], bvv = bv[d];
    const int half8 = (lane & 16) ? 8 : 0;
    for (int v = 0; v < 8; ++v) {
        const int r = rt * 16 + v + half8;
        const int p = r >> 10, n = r & 1023;
        const size_t qk = ((size_t)(m * P_ + p) * N_ + n) * C_ + c;
        Qh[qk] = (_Float16)((aq[v] + bqv) * 0.125f);
        Kh[qk] = (_Float16)(ak[v] + bkv);
        Vt[((size_t)(m * P_ + p) * C_ + c) * N_ + n] = (_Float16)(av[v] + bvv);
    }
}

// ---------------- flash attention ----------------
// grid = (N/128, M*P), block = 256 (8 waves), wave w owns queries [128*bx + 16w, +16).
// S^T = K * Q^T so score tile has lane=query, VGPR=key (cheap softmax + lane-local repack
// of exp(S) into the A fragment of P*V). K/V chunks double-buffered via async copies.
__global__ __launch_bounds__(256) void flash_attn_kernel(
    const _Float16* __restrict__ Qh, const _Float16* __restrict__ Kh,
    const _Float16* __restrict__ Vt, _Float16* __restrict__ attn_out)
{
    __shared__ __align__(16) _Float16 ldsK[2][128 * C_];   // [key][c]  2 x 16 KB
    __shared__ __align__(16) _Float16 ldsV[2][C_ * 128];   // [c][key]  2 x 16 KB
    const int mp = blockIdx.y;
    const int m = mp >> 3, p = mp & 7;
    const int wave = threadIdx.x >> 5;
    const int lane = threadIdx.x & 31;
    const int half8 = (lane & 16) ? 8 : 0;
    const int q0 = blockIdx.x * 128 + wave * 16;
    const _Float16* Qb = Qh + (size_t)mp * N_ * C_;
    const _Float16* Kb = Kh + (size_t)mp * N_ * C_;
    const _Float16* Vb = Vt + (size_t)mp * C_ * N_;

    // issue 8 async b128 copies (4 K + 4 V) for chunk `ch` into buffer buf
    auto stage = [&](int ch, int buf) {
        const int k0g = ch * 128;
        const uint4* srcK = reinterpret_cast<const uint4*>(Kb + (size_t)k0g * C_);
        uint4* dK = reinterpret_cast<uint4*>(&ldsK[buf][0]);
        uint4* dV = reinterpret_cast<uint4*>(&ldsV[buf][0]);
        for (int i = 0; i < 4; ++i) {
            const int idx = threadIdx.x + i * 256;
            async_copy_b128(srcK + idx, dK + idx);
            const int r = idx >> 4, off = idx & 15;
            async_copy_b128(reinterpret_cast<const uint4*>(Vb + (size_t)r * N_ + k0g) + off,
                            dV + idx);
        }
    };

    // Q as B operand of S^T = K * Q^T (K-dim = c = 64 -> two fragments, live whole loop)
    v16h qf0 = load_frag_b(Qb + (size_t)q0 * C_ + 0,  C_, lane);
    v16h qf1 = load_frag_b(Qb + (size_t)q0 * C_ + 32, C_, lane);

    float mrun = -__builtin_inff();
    float lrun = 0.f;
    v8f o[4] = {zero8(), zero8(), zero8(), zero8()};

    stage(0, 0);
    for (int ch = 0; ch < 8; ++ch) {
        const int buf = ch & 1;
        if (ch < 7) {
            stage(ch + 1, buf ^ 1);   // prefetch next chunk into other buffer
            wait_async_le8();         // async loads retire in order: current chunk complete
        } else {
            wait_async_0();
        }
        __syncthreads();

        const _Float16* lK = &ldsK[buf][0];
        const _Float16* lV = &ldsV[buf][0];

        // S^T tiles: 8 key-tiles x (2 WMMA over c)
        v8f s[8];
        for (int kt = 0; kt < 8; ++kt) {
            v16h ka0 = load_frag_a(lK + kt * 16 * C_ + 0,  C_, lane);
            v16h ka1 = load_frag_a(lK + kt * 16 * C_ + 32, C_, lane);
            v8f acc = zero8();
            acc = wmma_f16(ka0, qf0, acc);
            acc = wmma_f16(ka1, qf1, acc);
            s[kt] = acc;
        }
        // online softmax (lane pair l, l+16 share query l&15, disjoint keys)
        float mx = -__builtin_inff();
        for (int kt = 0; kt < 8; ++kt)
            for (int v = 0; v < 8; ++v) mx = fmaxf(mx, s[kt][v]);
        mx = fmaxf(mx, __shfl_xor(mx, 16, 32));
        const float mnew = fmaxf(mrun, mx);
        const float corr = exp2f((mrun - mnew) * LOG2E);
        float psum = 0.f;
        for (int kt = 0; kt < 8; ++kt)
            for (int v = 0; v < 8; ++v) {
                float pe = exp2f((s[kt][v] - mnew) * LOG2E);
                s[kt][v] = pe;
                psum += pe;
            }
        psum += __shfl_xor(psum, 16, 32);
        lrun = lrun * corr + psum;
        mrun = mnew;
        // rescale O accumulators (per-query factor broadcast to C-layout rows)
        for (int v = 0; v < 8; ++v) {
            const float cf = __shfl(corr, v + half8, 32);
            o[0][v] *= cf; o[1][v] *= cf; o[2][v] *= cf; o[3][v] *= cf;
        }
        // P (16q x 32k) A fragments: lane-local f32->f16 pack of two S^T tiles
        for (int kp = 0; kp < 4; ++kp) {
            union { v16h h; _Float16 e[16]; } af;
            for (int j = 0; j < 8; ++j) {
                af.e[j]     = (_Float16)s[2 * kp][j];
                af.e[8 + j] = (_Float16)s[2 * kp + 1][j];
            }
            for (int ct = 0; ct < 4; ++ct) {
                v16h vb = load_frag_b(lV + (ct * 16) * 128 + kp * 32, 128, lane);
                o[ct] = wmma_f16(af.h, vb, o[ct]);
            }
        }
        __syncthreads();   // block done with buf before it is refilled at ch+2
    }
    // normalize and store merged-heads (feature = m*C + c), f16 for next GEMM
    const float linv = 1.0f / lrun;
    for (int v = 0; v < 8; ++v) {
        const float li = __shfl(linv, v + half8, 32);
        o[0][v] *= li; o[1][v] *= li; o[2][v] *= li; o[3][v] *= li;
    }
    const int cl = lane & 15;
    for (int ct = 0; ct < 4; ++ct) {
        const int feat = m * C_ + ct * 16 + cl;
        for (int v = 0; v < 8; ++v) {
            const int q = q0 + v + half8;
            attn_out[((size_t)(p * N_ + q)) * D_ + feat] = (_Float16)o[ct][v];
        }
    }
}

// ---------------- generic GEMM + bias (+ReLU), out = A[R,K] @ W[Nout,K]^T + b ----------------
// 8 waves per block share one 16-row A tile (async-staged in LDS); each wave does 16x64 output.
__global__ __launch_bounds__(256) void gemm_bias_kernel(
    const _Float16* __restrict__ A, const _Float16* __restrict__ W,
    const float* __restrict__ bias, void* __restrict__ out,
    int K, int Nout, int relu, int f16out)
{
    __shared__ __align__(16) _Float16 ldsA[16 * 1024];   // up to K=1024 (32 KB)
    const int wid  = blockIdx.x * 8 + (threadIdx.x >> 5);
    const int lane = threadIdx.x & 31;
    const int half8 = (lane & 16) ? 8 : 0;
    const int nt = Nout >> 6;            // 8 or 16 -> rt uniform within block
    const int rt = wid / nt;
    const int cbase = (wid % nt) * 64;

    // stage shared A tile: 16 rows x K f16 = K/128 uint4 per thread
    {
        const uint4* src = reinterpret_cast<const uint4*>(A + (size_t)rt * 16 * K);
        uint4* dst = reinterpret_cast<uint4*>(ldsA);
        const int per = K >> 7;          // 4 (K=512) or 8 (K=1024)
        for (int i = 0; i < per; ++i) {
            const int idx = threadIdx.x + i * 256;
            async_copy_b128(src + idx, dst + idx);
        }
    }
    wait_async_0();
    __syncthreads();

    v8f acc[4] = {zero8(), zero8(), zero8(), zero8()};
    for (int k0 = 0; k0 < K; k0 += 32) {
        v16h a = load_frag_a(ldsA + k0, K, lane);
        for (int j = 0; j < 4; ++j) {
            v16h b = load_frag_b(W + (size_t)(cbase + j * 16) * K + k0, K, lane);
            acc[j] = wmma_f16(a, b, acc[j]);
        }
    }
    for (int j = 0; j < 4; ++j) {
        const int d = cbase + j * 16 + (lane & 15);
        const float bv = bias[d];
        for (int v = 0; v < 8; ++v) {
            float val = acc[j][v] + bv;
            if (relu) val = fmaxf(val, 0.f);
            const int r = rt * 16 + v + half8;
            if (f16out) ((_Float16*)out)[(size_t)r * Nout + d] = (_Float16)val;
            else        ((float*)out)[(size_t)r * Nout + d] = val;
        }
    }
}

extern "C" void kernel_launch(void* const* d_in, const int* in_sizes, int n_in,
                              void* d_out, int out_size, void* d_ws, size_t ws_size,
                              hipStream_t stream) {
    const float* h  = (const float*)d_in[0];
    const float* Wq = (const float*)d_in[1];
    const float* bq = (const float*)d_in[2];
    const float* Wk = (const float*)d_in[3];
    const float* bk = (const float*)d_in[4];
    const float* Wv = (const float*)d_in[5];
    const float* bv = (const float*)d_in[6];
    const float* Wo = (const float*)d_in[7];
    const float* bo = (const float*)d_in[8];
    const float* W1 = (const float*)d_in[9];
    const float* b1 = (const float*)d_in[10];
    const float* W2 = (const float*)d_in[11];
    const float* b2 = (const float*)d_in[12];
    float* out = (float*)d_out;

    char* ws = (char*)d_ws;
    size_t off = 0;
    auto alloc = [&](size_t bytes) -> _Float16* {
        _Float16* p = (_Float16*)(ws + off);
        off += (bytes + 255) & ~(size_t)255;
        return p;
    };
    _Float16* h16   = alloc((size_t)PN_ * D_ * 2);       // 8 MB
    _Float16* Wq16  = alloc((size_t)D_ * D_ * 2);
    _Float16* Wk16  = alloc((size_t)D_ * D_ * 2);
    _Float16* Wv16  = alloc((size_t)D_ * D_ * 2);
    _Float16* Wo16  = alloc((size_t)D_ * D_ * 2);
    _Float16* W116  = alloc((size_t)2 * D_ * D_ * 2);
    _Float16* W216  = alloc((size_t)2 * D_ * D_ * 2);
    _Float16* Qh16  = alloc((size_t)PN_ * D_ * 2);       // [m,p,n,c]
    _Float16* Kh16  = alloc((size_t)PN_ * D_ * 2);
    _Float16* Vt16  = alloc((size_t)PN_ * D_ * 2);       // [m,p,c,n]
    _Float16* at16  = alloc((size_t)PN_ * D_ * 2);       // merged-head attention out
    _Float16* y16   = alloc((size_t)PN_ * D_ * 2);       // after Wo
    _Float16* z16   = alloc((size_t)PN_ * 2 * D_ * 2);   // after W1+ReLU (16 MB)

    auto cvt = [&](const float* s, _Float16* d, int n) {
        cvt_f32_f16_kernel<<<(n + 255) / 256, 256, 0, stream>>>(s, d, n);
    };
    cvt(h,  h16,  PN_ * D_);
    cvt(Wq, Wq16, D_ * D_);
    cvt(Wk, Wk16, D_ * D_);
    cvt(Wv, Wv16, D_ * D_);
    cvt(Wo, Wo16, D_ * D_);
    cvt(W1, W116, 2 * D_ * D_);
    cvt(W2, W216, 2 * D_ * D_);

    // QKV: (PN/16)*(D/16) = 16384 wave-tiles / 8 waves = 2048 blocks
    qkv_kernel<<<2048, 256, 0, stream>>>(h16, Wq16, Wk16, Wv16, bq, bk, bv,
                                         Qh16, Kh16, Vt16);

    // flash attention: 8 query-blocks x 64 (m,p) pairs
    flash_attn_kernel<<<dim3(N_ / 128, M_ * P_), 256, 0, stream>>>(Qh16, Kh16, Vt16, at16);

    // Wo projection: (8192/16)*(512/64)/8 = 512 blocks
    gemm_bias_kernel<<<512, 256, 0, stream>>>(at16, Wo16, bo, (void*)y16,
                                              D_, D_, /*relu=*/0, /*f16out=*/1);
    // FFN up + ReLU: (8192/16)*(1024/64)/8 = 1024 blocks
    gemm_bias_kernel<<<1024, 256, 0, stream>>>(y16, W116, b1, (void*)z16,
                                               D_, 2 * D_, /*relu=*/1, /*f16out=*/1);
    // FFN down -> f32 output: 512 blocks
    gemm_bias_kernel<<<512, 256, 0, stream>>>(z16, W216, b2, (void*)out,
                                              2 * D_, D_, /*relu=*/0, /*f16out=*/0);
}
